// Model_60352880443415
// MI455X (gfx1250) — matrix operand order repaired and checked by reading, compiled, not run on hardware
//
#include <hip/hip_runtime.h>

// CRF forward (rescaled linear-domain scan via WMMA) + gold score.
// B=512 sequences, T=2048, K=32 tags. wave32: 16 sequences per wave,
// transition contraction = 2x v_wmma_f32_16x16x32_bf16 per time step.

#define KTAGS     32
#define TLEN      2048
#define NBATCH    512
#define START_TAG 30
#define STOP_TAG  31

typedef __attribute__((ext_vector_type(16))) __bf16 v16bf;
typedef __attribute__((ext_vector_type(8)))  float  v8f;
typedef __attribute__((ext_vector_type(4)))  float  v4f;

__global__ __launch_bounds__(128) void crf_forward_wmma(
    const float* __restrict__ feats,
    const float* __restrict__ trans,
    float* __restrict__ fwd_out)
{
  const int lane = threadIdx.x & 31;
  const int wave = blockIdx.x * (blockDim.x >> 5) + (threadIdx.x >> 5);
  const int b0   = wave << 4;        // 16 sequences per wave
  const int h    = lane >> 4;        // lane half (K/M split in WMMA layouts)
  const int n    = lane & 15;        // B/D column (sequence) id; A row id

  // ---- Preload E = exp(trans) in A-operand layout (bf16 16x32, 2 M-tiles) ----
  // A vgpr v, lane (h, m=lane%16): K pair = 16h + 2v
  v16bf A0, A1;
  #pragma unroll
  for (int v = 0; v < 8; ++v) {
    const int k0 = 2 * v + 16 * h;
    A0[2*v]   = (__bf16)__expf(trans[n * KTAGS + k0]);
    A0[2*v+1] = (__bf16)__expf(trans[n * KTAGS + k0 + 1]);
    A1[2*v]   = (__bf16)__expf(trans[(16 + n) * KTAGS + k0]);
    A1[2*v+1] = (__bf16)__expf(trans[(16 + n) * KTAGS + k0 + 1]);
  }

  // ---- State a[k][seq] in B-operand layout (bf16 32x16). Init: one-hot START.
  // B vgpr v, lane (h, n): K pair = 16h + 2v. K=30 -> h=1, v=7, lo element.
  v16bf Bop = {};
  if (h == 1) Bop[14] = (__bf16)1.0f;

  const float* fp = feats + (size_t)(b0 + n) * TLEN * KTAGS;
  float logZ = 0.0f;
  float rcp  = 1.0f;
  v8f D0 = {}, D1 = {};

  for (int t = 0; t < TLEN; ++t) {
    const float* e = fp + t * KTAGS;
    __builtin_prefetch(e + 8 * KTAGS, 0, 0);   // global_prefetch_b8, 8 steps ahead

    // D[m][n] = sum_k E[m][k] * a[k][n]   (2 output tag tiles)
    v8f zero = {};
    D0 = __builtin_amdgcn_wmma_f32_16x16x32_bf16(false, A0, false, Bop,
                                                 (short)0, zero, false, false);
    D1 = __builtin_amdgcn_wmma_f32_16x16x32_bf16(false, A1, false, Bop,
                                                 (short)0, zero, false, false);

    // Emission multiply: D tile0 vgpr r <-> tag m = r + 8h ; tile1 <-> 16 + r + 8h
    v4f f0a = *(const v4f*)(e + 8 * h);
    v4f f0b = *(const v4f*)(e + 8 * h + 4);
    v4f f1a = *(const v4f*)(e + 16 + 8 * h);
    v4f f1b = *(const v4f*)(e + 16 + 8 * h + 4);
    #pragma unroll
    for (int r = 0; r < 4; ++r) {
      D0[r]     *= __expf(f0a[r]);
      D0[r + 4] *= __expf(f0b[r]);
      D1[r]     *= __expf(f1a[r]);
      D1[r + 4] *= __expf(f1b[r]);
    }

    // Per-sequence normalizer c[n] = max_m D[m][n] (16 in-lane + one half-swap)
    float c = D0[0];
    #pragma unroll
    for (int r = 1; r < 8; ++r) c = fmaxf(c, D0[r]);
    #pragma unroll
    for (int r = 0; r < 8; ++r) c = fmaxf(c, D1[r]);
    c = fmaxf(c, __shfl_xor(c, 16, 32));
    rcp = 1.0f / c;
    logZ += __logf(c);

    // Rebuild B operand for next step: a[k][n] = D * rcp, bf16.
    // Needed K at lane (h,n): 16h+2v (+1). Source tile = h; cross-half swap:
    //   v<4 : lanes>=16 pull tile1 from lane^16 ; v>=4: lanes<16 pull tile0.
    v16bf nB;
    #pragma unroll
    for (int v = 0; v < 8; ++v) {
      float lo, hi;
      if (v < 4) {
        float slo = __shfl_xor(D1[2 * v], 16, 32);
        float shi = __shfl_xor(D1[2 * v + 1], 16, 32);
        lo = (h == 0) ? D0[2 * v]     : slo;
        hi = (h == 0) ? D0[2 * v + 1] : shi;
      } else {
        float slo = __shfl_xor(D0[2 * v - 8], 16, 32);
        float shi = __shfl_xor(D0[2 * v - 7], 16, 32);
        lo = (h == 0) ? slo : D1[2 * v - 8];
        hi = (h == 0) ? shi : D1[2 * v - 7];
      }
      nB[2 * v]     = (__bf16)(lo * rcp);
      nB[2 * v + 1] = (__bf16)(hi * rcp);
    }
    Bop = nB;
  }

  // ---- Termination: fwd = logZ + log( sum_m a_T[m][n] * exp(trans[STOP][m]) )
  float partial = 0.0f;
  #pragma unroll
  for (int r = 0; r < 8; ++r) {
    partial += D0[r] * rcp * __expf(trans[STOP_TAG * KTAGS + (r + 8 * h)]);
    partial += D1[r] * rcp * __expf(trans[STOP_TAG * KTAGS + (16 + r + 8 * h)]);
  }
  partial += __shfl_xor(partial, 16, 32);
  const float fwd = logZ + __logf(partial);
  if (h == 0) fwd_out[b0 + n] = fwd;
}

// Gold path score + final loss: one wave per sequence.
__global__ __launch_bounds__(256) void crf_gold(
    const float* __restrict__ feats,
    const float* __restrict__ trans,
    const int*   __restrict__ tags,
    const float* __restrict__ fwd,
    float*       __restrict__ out)
{
  const int lane = threadIdx.x & 31;
  const int b = blockIdx.x * (blockDim.x >> 5) + (threadIdx.x >> 5);
  if (b >= NBATCH) return;

  const int*   tg = tags  + (size_t)b * TLEN;
  const float* fb = feats + (size_t)b * TLEN * KTAGS;

  float s = 0.0f;
  for (int t = lane; t < TLEN; t += 32) {
    const int cur  = tg[t];
    const int prev = (t == 0) ? START_TAG : tg[t - 1];
    s += trans[cur * KTAGS + prev] + fb[(size_t)t * KTAGS + cur];
  }
  #pragma unroll
  for (int off = 16; off > 0; off >>= 1) s += __shfl_xor(s, off, 32);

  if (lane == 0) {
    s += trans[STOP_TAG * KTAGS + tg[TLEN - 1]];
    out[b] = fwd[b] - s;
  }
}

extern "C" void kernel_launch(void* const* d_in, const int* in_sizes, int n_in,
                              void* d_out, int out_size, void* d_ws, size_t ws_size,
                              hipStream_t stream) {
  const float* feats = (const float*)d_in[0];   // [512, 2048, 32] f32
  const float* trans = (const float*)d_in[1];   // [32, 32] f32
  const int*   tags  = (const int*)d_in[2];     // [512, 2048] int
  float* out    = (float*)d_out;                // [512] f32 losses
  float* fwd_ws = (float*)d_ws;                 // [512] f32 forward scores

  // 512 sequences / 16 per wave = 32 waves; 4 waves per 128-thread block.
  crf_forward_wmma<<<8, 128, 0, stream>>>(feats, trans, fwd_ws);
  // 512 sequences, 8 waves per 256-thread block.
  crf_gold<<<64, 256, 0, stream>>>(feats, trans, tags, fwd_ws, out);
}
